// MoELayer_64287070486630
// MI455X (gfx1250) — matrix operand order repaired and checked by reading, not compiled, not run on hardware
//
#include <hip/hip_runtime.h>
#include <math.h>

// Problem constants (from reference setup_inputs)
#define T_TOK 4096
#define H_DIM 1024
#define I_DIM 2048
#define E_NUM 8
#define EPSV 1e-6f

// FFN1: 128M x 64N block tiles
#define MT_A (T_TOK / 128)  // 32 max M-tiles per expert
#define NT_1 (I_DIM / 64)   // 32 N-tiles
// FFN2: 128M x 128N block tiles
#define NT_2 (H_DIM / 128)  // 8 N-tiles

typedef __attribute__((ext_vector_type(16))) __bf16 v16bf;
typedef __attribute__((ext_vector_type(8)))  float  v8f;
typedef unsigned int u32x4 __attribute__((ext_vector_type(4)));

union FragU { v16bf v; u32x4 q[2]; };

__device__ __forceinline__ unsigned short f2bf(float f) {
  union { float f; unsigned u; } x; x.f = f;
  unsigned u = x.u;
  u += 0x7fffu + ((u >> 16) & 1u);       // round-to-nearest-even
  return (unsigned short)(u >> 16);
}

// Fast sigmoid for the FFN1 epilogue: v_exp_f32 + v_rcp_f32.
__device__ __forceinline__ float fast_sigmoid(float v) {
  return __builtin_amdgcn_rcpf(1.0f + __expf(-v));
}

// A-matrix 16x32 bf16 fragment (ISA 7.12.2): lane l holds row M=l%16,
// halves [K: c..c+7] and [K: 16+c..16+c+7] with c = (l/16)*8.
__device__ __forceinline__ v16bf load_frag_A(const unsigned short* rowptr, int lane, int k0) {
  int c = (lane >> 4) << 3;
  FragU f;
  f.q[0] = *reinterpret_cast<const u32x4*>(rowptr + k0 + c);
  f.q[1] = *reinterpret_cast<const u32x4*>(rowptr + k0 + c + 16);
  return f.v;
}

// B-matrix 32x16 bf16 fragment from N-major storage (row = N, contiguous K):
// lane l holds col N=l%16, halves [K: c..c+7] and [K: 16+c..16+c+7] with c = (l/16)*8.
__device__ __forceinline__ v16bf load_frag_B(const unsigned short* rowptr, int lane, int k0) {
  int c = (lane >> 4) << 3;
  FragU f;
  f.q[0] = *reinterpret_cast<const u32x4*>(rowptr + k0 + c);
  f.q[1] = *reinterpret_cast<const u32x4*>(rowptr + k0 + c + 16);
  return f.v;
}

__device__ __forceinline__ v8f wmma_bf16(v16bf a, v16bf b, v8f c) {
  return __builtin_amdgcn_wmma_f32_16x16x32_bf16(false, a, false, b, (short)0, c, false, false);
}

// ---------------------------------------------------------------- init
__global__ void init_kernel(float* __restrict__ out, size_t out_n,
                            int* __restrict__ cnt, int* __restrict__ cnt2,
                            int* __restrict__ basep, float* __restrict__ aux) {
  size_t i = (size_t)blockIdx.x * blockDim.x + threadIdx.x;
  size_t stride = (size_t)gridDim.x * blockDim.x;
  for (size_t j = i; j < out_n; j += stride) out[j] = 0.0f;
  if (blockIdx.x == 0 && threadIdx.x < 16) {
    if (threadIdx.x < E_NUM) {
      cnt[threadIdx.x] = 0; cnt2[threadIdx.x] = 0; basep[threadIdx.x] = 0;
    }
    if (threadIdx.x < E_NUM + 2) aux[threadIdx.x] = 0.0f;  // [0..7]=probsum [8]=z [9]=ent
  }
}

// ---------------------------------------------------------------- cast x
__global__ void cast_f32_bf16(const float* __restrict__ in, unsigned short* __restrict__ out, size_t n) {
  size_t i = (size_t)blockIdx.x * blockDim.x + threadIdx.x;
  size_t stride = (size_t)gridDim.x * blockDim.x;
  for (size_t j = i; j < n; j += stride) out[j] = f2bf(in[j]);
}

// ------------------------------------------------- transpose + cast (LDS tiled)
// in: [batch][R][C] f32  ->  out: [batch][C][R] bf16
__global__ __launch_bounds__(256) void transpose_cast(const float* __restrict__ in,
                                                      unsigned short* __restrict__ out,
                                                      int R, int C) {
  __shared__ float tile[32][33];
  size_t bo = (size_t)blockIdx.z * R * C;
  int r0 = blockIdx.y * 32, c0 = blockIdx.x * 32;
  int tx = threadIdx.x, ty = threadIdx.y;
  #pragma unroll
  for (int i = 0; i < 32; i += 8) {
    tile[ty + i][tx] = in[bo + (size_t)(r0 + ty + i) * C + (c0 + tx)];
  }
  __syncthreads();
  #pragma unroll
  for (int i = 0; i < 32; i += 8) {
    out[bo + (size_t)(c0 + ty + i) * R + (r0 + tx)] = f2bf(tile[tx][ty + i]);
  }
}

// ---------------------------------------------------------------- router
__global__ __launch_bounds__(256) void router_kernel(const float* __restrict__ x,
                                                     const float* __restrict__ rw,
                                                     int* __restrict__ tk_i, float* __restrict__ tk_w,
                                                     int* __restrict__ cnt, float* __restrict__ aux) {
  __shared__ float s_rw[E_NUM * H_DIM];     // 32 KB
  __shared__ float s_red[E_NUM + 2];
  for (int i = threadIdx.x; i < E_NUM * H_DIM; i += blockDim.x) s_rw[i] = rw[i];
  if (threadIdx.x < E_NUM + 2) s_red[threadIdx.x] = 0.0f;
  __syncthreads();

  int t = blockIdx.x * blockDim.x + threadIdx.x;   // T divisible by 256
  const float* xr = x + (size_t)t * H_DIM;
  float acc[E_NUM];
  #pragma unroll
  for (int e = 0; e < E_NUM; ++e) acc[e] = 0.0f;
  for (int h = 0; h < H_DIM; ++h) {
    float xv = xr[h];
    #pragma unroll
    for (int e = 0; e < E_NUM; ++e) acc[e] = fmaf(xv, s_rw[e * H_DIM + h], acc[e]);
  }
  float m = -1e30f;
  #pragma unroll
  for (int e = 0; e < E_NUM; ++e) {
    acc[e] = fminf(fmaxf(acc[e], -50.0f), 50.0f);
    m = fmaxf(m, acc[e]);
  }
  float sum = 0.0f, p[E_NUM];
  #pragma unroll
  for (int e = 0; e < E_NUM; ++e) { p[e] = expf(acc[e] - m); sum += p[e]; }
  float inv = 1.0f / sum;
  float lse = m + logf(sum);
  float ent = 0.0f;
  #pragma unroll
  for (int e = 0; e < E_NUM; ++e) {
    p[e] *= inv;
    float ps = fminf(fmaxf(p[e], EPSV), 1.0f);
    ent -= ps * logf(ps);
    atomicAdd(&s_red[e], p[e]);
  }
  atomicAdd(&s_red[E_NUM + 0], lse * lse);
  atomicAdd(&s_red[E_NUM + 1], ent);

  // top-2 (first index wins ties, matching lax.top_k)
  int i1 = 0; float p1 = p[0];
  #pragma unroll
  for (int e = 1; e < E_NUM; ++e) if (p[e] > p1) { p1 = p[e]; i1 = e; }
  int i2 = (i1 == 0) ? 1 : 0; float p2 = p[i2];
  #pragma unroll
  for (int e = 0; e < E_NUM; ++e) if (e != i1 && p[e] > p2) { p2 = p[e]; i2 = e; }
  float denom = 1.0f / (p1 + p2 + EPSV);
  tk_i[t * 2 + 0] = i1; tk_i[t * 2 + 1] = i2;
  tk_w[t * 2 + 0] = p1 * denom; tk_w[t * 2 + 1] = p2 * denom;
  atomicAdd(&cnt[i1], 1);
  atomicAdd(&cnt[i2], 1);

  __syncthreads();
  if (threadIdx.x < E_NUM + 2) atomicAdd(&aux[threadIdx.x], s_red[threadIdx.x]);
}

__global__ void prefix_kernel(const int* __restrict__ cnt, int* __restrict__ basep) {
  if (blockIdx.x == 0 && threadIdx.x == 0) {
    int b = 0;
    for (int e = 0; e < E_NUM; ++e) { basep[e] = b; b += cnt[e]; }
  }
}

__global__ void scatter_kernel(const int* __restrict__ tk_i, const float* __restrict__ tk_w,
                               const int* __restrict__ basep, int* __restrict__ cnt2,
                               int* __restrict__ slot_tok, float* __restrict__ slot_w) {
  int t = blockIdx.x * blockDim.x + threadIdx.x;
  if (t >= T_TOK) return;
  #pragma unroll
  for (int k = 0; k < 2; ++k) {
    int e = tk_i[t * 2 + k];
    int pos = atomicAdd(&cnt2[e], 1);
    int s = basep[e] + pos;
    slot_tok[s] = t;
    slot_w[s] = tk_w[t * 2 + k];
  }
}

// --------------------------------------------- FFN1: h = silu(x@wg)*(x@wu)*cw
// 128M x 64N block tile; wave = 4 M-subtiles x 1 N-subtile x {gate,up}
// -> 8 v_wmma per 12 global_load_b128 per K-step.
// e in [0..E_NUM): routed experts; e == E_NUM: shared expert (cw = sigmoid(gate))
__global__ __launch_bounds__(256) void ffn1_kernel(
    const unsigned short* __restrict__ xb,
    const unsigned short* __restrict__ wgt, const unsigned short* __restrict__ wut,   // [E][I][H] N-major
    const unsigned short* __restrict__ swgt, const unsigned short* __restrict__ swut, // [I][H]
    const int* __restrict__ cnt, const int* __restrict__ basep,
    const int* __restrict__ slot_tok, const float* __restrict__ slot_w,
    const float* __restrict__ shared_gate,
    unsigned short* __restrict__ hbuf,   // [2T][I] slot-major
    unsigned short* __restrict__ hs) {   // [T][I]
  const int bpe = MT_A * NT_1;
  int e = blockIdx.x / bpe;
  int rem = blockIdx.x % bpe;
  int mt = rem / NT_1, nt = rem % NT_1;

  __shared__ int   s_tok[128];
  __shared__ float s_w[128];

  int rows, rowbase = 0;
  const unsigned short *Bg, *Bu;
  unsigned short* hout;
  if (e < E_NUM) {
    rows = cnt[e];
    if (mt * 128 >= rows) return;
    rowbase = basep[e];
    Bg = wgt + (size_t)e * I_DIM * H_DIM;
    Bu = wut + (size_t)e * I_DIM * H_DIM;
    hout = hbuf;
  } else {
    rows = T_TOK;
    Bg = swgt; Bu = swut;
    hout = hs;
  }

  int tid = threadIdx.x;
  if (tid < 128) {
    int s = mt * 128 + tid;
    if (e < E_NUM) {
      bool v = s < rows;
      s_tok[tid] = v ? slot_tok[rowbase + s] : 0;
      s_w[tid]   = v ? slot_w[rowbase + s] : 0.0f;
    } else {
      s_tok[tid] = s;
      float g0 = shared_gate[0];
      s_w[tid] = 1.0f / (1.0f + expf(-g0));
    }
  }
  __syncthreads();

  int lane = tid & 31, w = tid >> 5;
  int ni = (w & 3) * 16;        // N-subtile offset within 64
  int mg = (w >> 2) * 64;       // M-group: 4 subtiles (64 rows)
  int lr = lane & 15;

  const unsigned short* ap[4];
  #pragma unroll
  for (int j = 0; j < 4; ++j)
    ap[j] = xb + (size_t)s_tok[mg + j * 16 + lr] * H_DIM;
  int ncol = nt * 64 + ni + lr;
  const unsigned short* bgp = Bg + (size_t)ncol * H_DIM;
  const unsigned short* bup = Bu + (size_t)ncol * H_DIM;

  v8f cg[4] = {{}, {}, {}, {}};
  v8f cu[4] = {{}, {}, {}, {}};
  for (int k0 = 0; k0 < H_DIM; k0 += 32) {
    v16bf a[4];
    #pragma unroll
    for (int j = 0; j < 4; ++j) a[j] = load_frag_A(ap[j], lane, k0);
    v16bf bg = load_frag_B(bgp, lane, k0);
    v16bf bu = load_frag_B(bup, lane, k0);
    #pragma unroll
    for (int j = 0; j < 4; ++j) {
      cg[j] = wmma_bf16(a[j], bg, cg[j]);
      cu[j] = wmma_bf16(a[j], bu, cu[j]);
    }
  }

  int rhalf = (lane >> 4) * 8;   // C/D layout: lane row-half offset
  #pragma unroll
  for (int j = 0; j < 4; ++j) {
    #pragma unroll
    for (int r = 0; r < 8; ++r) {
      int rowt = mg + j * 16 + rhalf + r;
      int s = mt * 128 + rowt;
      if (s < rows) {
        float gv = cg[j][r], uv = cu[j][r];
        float hv = gv * fast_sigmoid(gv) * uv * s_w[rowt];
        size_t orow = (e < E_NUM) ? (size_t)(rowbase + s) : (size_t)s;
        hout[orow * I_DIM + ncol] = f2bf(hv);
      }
    }
  }
}

// --------------------------------------------- FFN2: out += h @ wd (scatter-add)
// 128M x 128N block tile; wave = 4 M-subtiles x 2 N-subtiles
// -> 8 v_wmma per 12 global_load_b128 per K-step.
__global__ __launch_bounds__(256) void ffn2_kernel(
    const unsigned short* __restrict__ hbuf, const unsigned short* __restrict__ hs,
    const unsigned short* __restrict__ wdt,  // [E][H][I] N-major
    const unsigned short* __restrict__ swdt, // [H][I]
    const int* __restrict__ cnt, const int* __restrict__ basep,
    const int* __restrict__ slot_tok, float* __restrict__ out) {
  const int bpe = MT_A * NT_2;
  int e = blockIdx.x / bpe;
  int rem = blockIdx.x % bpe;
  int mt = rem / NT_2, nt = rem % NT_2;

  __shared__ int s_tok[128];

  int rows, rowbase = 0;
  const unsigned short *Bp, *Ap;
  if (e < E_NUM) {
    rows = cnt[e];
    if (mt * 128 >= rows) return;
    rowbase = basep[e];
    Bp = wdt + (size_t)e * H_DIM * I_DIM;
    Ap = hbuf;
  } else {
    rows = T_TOK;
    Bp = swdt;
    Ap = hs;
  }

  int tid = threadIdx.x;
  if (tid < 128) {
    int s = mt * 128 + tid;
    if (e < E_NUM) s_tok[tid] = (s < rows) ? slot_tok[rowbase + s] : 0;
    else           s_tok[tid] = s;
  }
  __syncthreads();

  int lane = tid & 31, w = tid >> 5;
  int ni = (w & 3) * 32;        // 2 N-subtiles per wave
  int mg = (w >> 2) * 64;       // 4 M-subtiles per wave
  int lr = lane & 15;

  const unsigned short* ap[4];
  #pragma unroll
  for (int j = 0; j < 4; ++j) {
    int sA = mt * 128 + mg + j * 16 + lr;
    if (sA > rows - 1) sA = rows - 1;
    ap[j] = Ap + ((size_t)rowbase + sA) * I_DIM;
  }
  int ncol0 = nt * 128 + ni + lr;
  int ncol1 = ncol0 + 16;
  const unsigned short* bp0 = Bp + (size_t)ncol0 * I_DIM;
  const unsigned short* bp1 = Bp + (size_t)ncol1 * I_DIM;

  v8f c0[4] = {{}, {}, {}, {}};
  v8f c1[4] = {{}, {}, {}, {}};
  for (int k0 = 0; k0 < I_DIM; k0 += 32) {
    v16bf a[4];
    #pragma unroll
    for (int j = 0; j < 4; ++j) a[j] = load_frag_A(ap[j], lane, k0);
    v16bf b0 = load_frag_B(bp0, lane, k0);
    v16bf b1 = load_frag_B(bp1, lane, k0);
    #pragma unroll
    for (int j = 0; j < 4; ++j) {
      c0[j] = wmma_bf16(a[j], b0, c0[j]);
      c1[j] = wmma_bf16(a[j], b1, c1[j]);
    }
  }

  int rhalf = (lane >> 4) * 8;
  #pragma unroll
  for (int j = 0; j < 4; ++j) {
    #pragma unroll
    for (int r = 0; r < 8; ++r) {
      int rowt = mg + j * 16 + rhalf + r;
      int s = mt * 128 + rowt;
      if (s < rows) {
        int t = s_tok[rowt];
        atomicAdd(&out[(size_t)t * H_DIM + ncol0], c0[j][r]);
        atomicAdd(&out[(size_t)t * H_DIM + ncol1], c1[j][r]);
      }
    }
  }
}

// ---------------------------------------------------------------- aux
__global__ void aux_final_kernel(const int* __restrict__ cnt, const float* __restrict__ aux,
                                 float* __restrict__ out_aux) {
  if (blockIdx.x == 0 && threadIdx.x == 0) {
    float denom = (float)(T_TOK * 2);
    float lb = 0.0f, usage = 0.0f;
    for (int e = 0; e < E_NUM; ++e) {
      float tpe = (float)cnt[e] / denom;
      lb += tpe * (aux[e] / (float)T_TOK);
      usage += (tpe > 0.01f) ? 1.0f : 0.0f;
    }
    lb *= (float)E_NUM;
    float z    = (aux[E_NUM + 0] / (float)T_TOK) * 0.001f;
    float entl = (logf((float)E_NUM) - aux[E_NUM + 1] / (float)T_TOK) * 0.01f;
    float util = (1.0f - usage / (float)E_NUM) * 0.1f;
    float a = lb + z + entl + util;
    out_aux[0] = fminf(fmaxf(a, 0.0f), 100.0f);
  }
}

// ================================================================ launch
extern "C" void kernel_launch(void* const* d_in, const int* in_sizes, int n_in,
                              void* d_out, int out_size, void* d_ws, size_t ws_size,
                              hipStream_t stream) {
  (void)in_sizes; (void)n_in; (void)out_size; (void)ws_size;
  const float* x   = (const float*)d_in[0];   // [B,S,H] -> [T,H]
  const float* rw  = (const float*)d_in[1];   // [E,H]
  const float* wg  = (const float*)d_in[2];   // [E,H,I]
  const float* wu  = (const float*)d_in[3];   // [E,H,I]
  const float* wd  = (const float*)d_in[4];   // [E,I,H]
  const float* swg = (const float*)d_in[5];   // [H,I]
  const float* swu = (const float*)d_in[6];   // [H,I]
  const float* swd = (const float*)d_in[7];   // [I,H]
  const float* sgate = (const float*)d_in[8]; // [1]
  float* out = (float*)d_out;                 // T*H floats + 1 aux

  char* ws = (char*)d_ws;
  size_t off = 0;
  auto alloc = [&](size_t bytes) -> void* {
    void* p = (void*)(ws + off);
    off += (bytes + 255) & ~(size_t)255;
    return p;
  };
  unsigned short* xb   = (unsigned short*)alloc((size_t)T_TOK * H_DIM * 2);
  unsigned short* wgt  = (unsigned short*)alloc((size_t)E_NUM * H_DIM * I_DIM * 2);
  unsigned short* wut  = (unsigned short*)alloc((size_t)E_NUM * H_DIM * I_DIM * 2);
  unsigned short* wdt  = (unsigned short*)alloc((size_t)E_NUM * H_DIM * I_DIM * 2);
  unsigned short* swgt = (unsigned short*)alloc((size_t)H_DIM * I_DIM * 2);
  unsigned short* swut = (unsigned short*)alloc((size_t)H_DIM * I_DIM * 2);
  unsigned short* swdt = (unsigned short*)alloc((size_t)H_DIM * I_DIM * 2);
  unsigned short* hbuf = (unsigned short*)alloc((size_t)2 * T_TOK * I_DIM * 2);
  unsigned short* hs   = (unsigned short*)alloc((size_t)T_TOK * I_DIM * 2);
  int*   tk_i     = (int*)alloc((size_t)T_TOK * 2 * 4);
  float* tk_w     = (float*)alloc((size_t)T_TOK * 2 * 4);
  int*   slot_tok = (int*)alloc((size_t)2 * T_TOK * 4);
  float* slot_w   = (float*)alloc((size_t)2 * T_TOK * 4);
  int*   cnt   = (int*)alloc(256);
  int*   cnt2  = (int*)alloc(256);
  int*   basep = (int*)alloc(256);
  float* aux   = (float*)alloc(256);

  dim3 tb(32, 8);

  init_kernel<<<2048, 256, 0, stream>>>(out, (size_t)T_TOK * H_DIM, cnt, cnt2, basep, aux);
  cast_f32_bf16<<<4096, 256, 0, stream>>>(x, xb, (size_t)T_TOK * H_DIM);

  transpose_cast<<<dim3(I_DIM / 32, H_DIM / 32, E_NUM), tb, 0, stream>>>(wg, wgt, H_DIM, I_DIM);
  transpose_cast<<<dim3(I_DIM / 32, H_DIM / 32, E_NUM), tb, 0, stream>>>(wu, wut, H_DIM, I_DIM);
  transpose_cast<<<dim3(H_DIM / 32, I_DIM / 32, E_NUM), tb, 0, stream>>>(wd, wdt, I_DIM, H_DIM);
  transpose_cast<<<dim3(I_DIM / 32, H_DIM / 32, 1), tb, 0, stream>>>(swg, swgt, H_DIM, I_DIM);
  transpose_cast<<<dim3(I_DIM / 32, H_DIM / 32, 1), tb, 0, stream>>>(swu, swut, H_DIM, I_DIM);
  transpose_cast<<<dim3(H_DIM / 32, I_DIM / 32, 1), tb, 0, stream>>>(swd, swdt, I_DIM, H_DIM);

  router_kernel<<<T_TOK / 256, 256, 0, stream>>>(x, rw, tk_i, tk_w, cnt, aux);
  prefix_kernel<<<1, 1, 0, stream>>>(cnt, basep);
  scatter_kernel<<<T_TOK / 256, 256, 0, stream>>>(tk_i, tk_w, basep, cnt2, slot_tok, slot_w);

  ffn1_kernel<<<(E_NUM + 1) * MT_A * NT_1, 256, 0, stream>>>(
      xb, wgt, wut, swgt, swut, cnt, basep, slot_tok, slot_w, sgate, hbuf, hs);
  ffn2_kernel<<<(E_NUM + 1) * MT_A * NT_2, 256, 0, stream>>>(
      hbuf, hs, wdt, swdt, cnt, basep, slot_tok, out);

  aux_final_kernel<<<1, 32, 0, stream>>>(cnt, aux, out + (size_t)T_TOK * H_DIM);
}
